// DACE_29884382445872
// MI455X (gfx1250) — matrix operand order repaired and checked by reading, not compiled, not run on hardware
//
#include <hip/hip_runtime.h>

typedef __attribute__((ext_vector_type(16))) _Float16 v16h;
typedef __attribute__((ext_vector_type(8)))  float    v8f;

#define NB        2048
#define IDIM      128
#define NROWS     (NB * IDIM)   // 262144
#define KK        256
#define EE        64
#define ROWS_PER_BLOCK 128      // 8 waves * 16 rows

// Soft-quantization embedding:
//   w[row,k] = softmax_k( -|g| * (x[row] - anchor[k])^2 )
//   out[row,e] = sum_k w[row,k] * emb[k,e]
// GEMM part done with v_wmma_f32_16x16x32_f16; weights built per-lane in the
// exact A-fragment layout, embeddings pre-packed in LDS in the B-fragment layout.
__global__ __launch_bounds__(256) void softquant_wmma_kernel(
    const float* __restrict__ x,
    const float* __restrict__ anchors,
    const float* __restrict__ emb,
    const float* __restrict__ gamma,
    float* __restrict__ out)
{
    // B fragments: [chunk c=0..7][etile t=0..3][lane=0..31][e=0..15] f16 = 32 KB
    __shared__ _Float16 sB[8 * 4 * 32 * 16];
    __shared__ float    sA[KK];                 // anchors, 1 KB

    const int tid = threadIdx.x;

    // ---- stage anchors ----
    if (tid < KK) sA[tid] = anchors[tid];

    // ---- stage embeddings pre-packed into WMMA B-fragment layout ----
    // B 32x16 f16 fragment, this code's contiguous k order (same as A): lane (h=lane>>4, n=lane&15) element e holds
    // B[K = 16*h + e][N = n].  For chunk c, K_global = 32*c + 16*h + e,
    // column = 16*t + n.
    for (int i = 0; i < 64; ++i) {
        int idx  = tid + i * 256;               // 16384 halfs total
        int c    = idx >> 11;
        int t    = (idx >> 9) & 3;
        int lane = (idx >> 4) & 31;
        int e    = idx & 15;
        int k    = 32 * c + 16 * (lane >> 4) + e;
        int col  = 16 * t + (lane & 15);
        sB[idx]  = (_Float16)emb[k * EE + col];
    }
    __syncthreads();

    const int lane = tid & 31;
    const int wave = tid >> 5;
    const int h    = lane >> 4;          // half-wave id
    const int r16  = lane & 15;          // row-in-tile (A/M index) / column (B,N / D,N)
    const int rowBase = blockIdx.x * ROWS_PER_BLOCK + wave * 16;

    const float xv = x[rowBase + r16];   // both half-waves hold the same 16 rows
    const float g  = fabsf(gamma[0]);

    // ---- softmax denominator: each lane scans its half (128 anchors) ----
    const float* anc = &sA[128 * h];
    float dmin = 1e30f;                            // min d^2 <=> max of -g*d^2
    for (int k = 0; k < 128; ++k) {
        float d = xv - anc[k];
        dmin = fminf(dmin, d * d);
    }
    dmin = fminf(dmin, __shfl_xor(dmin, 16, 32));  // combine halves

    float sum = 0.0f;
    for (int k = 0; k < 128; ++k) {
        float d = xv - anc[k];
        sum += __expf(g * (dmin - d * d));
    }
    sum += __shfl_xor(sum, 16, 32);
    const float isum = 1.0f / sum;

    // ---- GEMM: 8 K-chunks of 32, 4 E-tiles of 16 ----
    v8f acc0 = {}, acc1 = {}, acc2 = {}, acc3 = {};
    const v16h* fB = (const v16h*)sB;              // v16h = one lane's fragment slice

    #pragma unroll
    for (int c = 0; c < 8; ++c) {
        // Build A fragment (16x32 f16): lane (h, M=r16) element e holds
        // A[M][K = 32*c + 16*h + e]
        v16h a;
        #pragma unroll
        for (int e = 0; e < 16; ++e) {
            int   kg = 32 * c + 16 * h + e;
            float d  = xv - sA[kg];
            a[e] = (_Float16)(__expf(g * (dmin - d * d)) * isum);
        }

        const v16h b0 = fB[(c * 4 + 0) * 32 + lane];
        const v16h b1 = fB[(c * 4 + 1) * 32 + lane];
        const v16h b2 = fB[(c * 4 + 2) * 32 + lane];
        const v16h b3 = fB[(c * 4 + 3) * 32 + lane];

        acc0 = __builtin_amdgcn_wmma_f32_16x16x32_f16(false, a, false, b0, (short)0, acc0, false, false);
        acc1 = __builtin_amdgcn_wmma_f32_16x16x32_f16(false, a, false, b1, (short)0, acc1, false, false);
        acc2 = __builtin_amdgcn_wmma_f32_16x16x32_f16(false, a, false, b2, (short)0, acc2, false, false);
        acc3 = __builtin_amdgcn_wmma_f32_16x16x32_f16(false, a, false, b3, (short)0, acc3, false, false);
    }

    // ---- store D tiles: lane N=r16, VGPR r -> M = 8*h + r ----
    #pragma unroll
    for (int r = 0; r < 8; ++r) {
        int row = rowBase + 8 * h + r;
        float* o = out + (long)row * EE + r16;
        o[0]  = acc0[r];
        o[16] = acc1[r];
        o[32] = acc2[r];
        o[48] = acc3[r];
    }
}

extern "C" void kernel_launch(void* const* d_in, const int* in_sizes, int n_in,
                              void* d_out, int out_size, void* d_ws, size_t ws_size,
                              hipStream_t stream) {
    const float* x       = (const float*)d_in[0];   // [2048,128]
    const float* anchors = (const float*)d_in[1];   // [256]
    const float* emb     = (const float*)d_in[2];   // [256,64]
    const float* gamma   = (const float*)d_in[3];   // scalar
    float*       out     = (float*)d_out;           // [2048, 8192]

    dim3 grid(NROWS / ROWS_PER_BLOCK);  // 2048
    dim3 block(256);                    // 8 wave32
    hipLaunchKernelGGL(softquant_wmma_kernel, grid, block, 0, stream,
                       x, anchors, emb, gamma, out);
}
